// SymForceField_41936060678666
// MI455X (gfx1250) — matrix operand order repaired and checked by reading, not compiled, not run on hardware
//
#include <hip/hip_runtime.h>
#include <hip/hip_bf16.h>

typedef __attribute__((ext_vector_type(16))) _Float16 v16h;
typedef __attribute__((ext_vector_type(8)))  _Float16 v8h;
typedef __attribute__((ext_vector_type(8)))  float    v8f;

#define NN 768
#define DD 64
#define SS 32
#define PP 4
#define NT 48   // 16-wide tiles per matrix dimension (768/16)

// ---- stage 0: hid (f32 [N,D]) -> f16 [N,D] (serves directly as WMMA B fragments) ----
__global__ void k_cvt_hid(const float* __restrict__ hid, _Float16* __restrict__ hid_h) {
    int i = blockIdx.x * 256 + threadIdx.x;
    if (i < NN * DD) hid_h[i] = (_Float16)hid[i];
}

// ---- stage 1: T[s,x,j] = sum_i hid[x,i] * Wsym[i,j,s], Wsym = (W + W^T)/2, f16 [S][N][D]
//      Symmetrizing W makes inter[x,y,s] symmetric in (x,y) against the symmetric RBF
//      weights, allowing upper-triangular tile enumeration in the fused kernel. ----
__global__ void k_calc_T(const float* __restrict__ hid, const float* __restrict__ W,
                         _Float16* __restrict__ T_h) {
    int tid = blockIdx.x * 256 + threadIdx.x;   // N*D*S threads; s fastest -> coalesced W reads
    int s = tid & (SS - 1);
    int j = (tid >> 5) & (DD - 1);
    int x = tid >> 11;
    const float* hrow = hid + x * DD;
    float a = 0.f;
    #pragma unroll 8
    for (int i = 0; i < DD; ++i) {
        float wsym = 0.5f * (W[(i * DD + j) * SS + s] + W[(j * DD + i) * SS + s]);
        a += hrow[i] * wsym;
    }
    T_h[(s * NN + x) * DD + j] = (_Float16)a;
}

__global__ void k_zero(float* acc) { if (threadIdx.x < PP) acc[threadIdx.x] = 0.f; }

// ---- stage 2+3 fused: upper-triangular 16x16 (x,y) tiles; WMMA over K=64 per s,
//      RBF-weighted reduction; off-diagonal tiles count twice by symmetry. ----
__global__ void __launch_bounds__(256)
k_fused(const _Float16* __restrict__ T_h, const _Float16* __restrict__ hid_h,
        const float* __restrict__ coords, const float* __restrict__ b_rbf,
        float* __restrict__ acc) {
    const int lane = threadIdx.x & 31;
    const int tile = blockIdx.x * 8 + (threadIdx.x >> 5);
    const int tx = tile / NT;
    const int ty = tile % NT;
    if (tx > ty) return;                 // wave-uniform: EXEC stays all-ones below
    const float tilew = (tx < ty) ? 2.0f : 1.0f;
    const int x0 = tx * 16;
    const int y0 = ty * 16;
    const int n  = lane & 15;
    const int hi = lane >> 4;

    // B fragments: B[k][y] = hid[y0+n][k]; lane n holds 16 contiguous K (hi selects K half)
    const _Float16* pb = hid_h + (y0 + n) * DD + hi * 16;
    v16h b0 = *(const v16h*)(pb);        // j = 0..31 chunk
    v16h b1 = *(const v16h*)(pb + 32);   // j = 32..63 chunk

    // C/D layout: element v of C is (x = x0 + v + 8*hi, y = y0 + n)
    float mask[8];
    #pragma unroll
    for (int v = 0; v < 8; ++v)
        mask[v] = (x0 + v + hi * 8 == y0 + n) ? 0.f : 1.f;

    float dist[PP][8];
    #pragma unroll
    for (int p = 0; p < PP; ++p) {
        const float* cp = coords + (size_t)p * NN * 3;
        float cyx = cp[(y0 + n) * 3 + 0];
        float cyy = cp[(y0 + n) * 3 + 1];
        float cyz = cp[(y0 + n) * 3 + 2];
        #pragma unroll
        for (int v = 0; v < 8; ++v) {
            int xr = x0 + v + hi * 8;
            float dx = cp[xr * 3 + 0] - cyx;
            float dy = cp[xr * 3 + 1] - cyy;
            float dz = cp[xr * 3 + 2] - cyz;
            dist[p][v] = sqrtf(dx * dx + dy * dy + dz * dz + 1e-12f);
        }
    }

    float accp[PP] = {0.f, 0.f, 0.f, 0.f};
    const float inv2s2 = 3.5555556f;     // 1/(2*sigma^2), sigma = 12/32
    const float dmu    = 12.0f / 31.0f;  // linspace(0,12,32) step

    #pragma unroll 2
    for (int s = 0; s < SS; ++s) {
        // A fragments of T_s: lane half hi holds 16 contiguous K (matches B)
        const _Float16* pa = T_h + ((size_t)(s * NN) + x0 + n) * DD + hi * 16;
        v8h a0lo = *(const v8h*)(pa);
        v8h a0hi = *(const v8h*)(pa + 8);
        v8h a1lo = *(const v8h*)(pa + 32);
        v8h a1hi = *(const v8h*)(pa + 40);
        v16h a0 = __builtin_shufflevector(a0lo, a0hi, 0,1,2,3,4,5,6,7,8,9,10,11,12,13,14,15);
        v16h a1 = __builtin_shufflevector(a1lo, a1hi, 0,1,2,3,4,5,6,7,8,9,10,11,12,13,14,15);

        v8f c = {};
        c = __builtin_amdgcn_wmma_f32_16x16x32_f16(false, a0, false, b0, (short)0, c, false, false);
        c = __builtin_amdgcn_wmma_f32_16x16x32_f16(false, a1, false, b1, (short)0, c, false, false);

        float br = b_rbf[s];
        float mu = (float)s * dmu;
        #pragma unroll
        for (int v = 0; v < 8; ++v) {
            float val = (c[v] + br) * mask[v];
            #pragma unroll
            for (int p = 0; p < PP; ++p) {
                float t = dist[p][v] - mu;
                accp[p] += val * __expf(-(t * t) * inv2s2);
            }
        }
    }

    #pragma unroll
    for (int p = 0; p < PP; ++p) {
        float a = accp[p] * tilew;
        #pragma unroll
        for (int off = 16; off > 0; off >>= 1) a += __shfl_xor(a, off, 32);
        if (lane == 0) atomicAdd(acc + p, a);
    }
}

// ---- stage 4: finalize U[p] = acc[p]/(N*N*S) * weight + bias ----
__global__ void k_final(const float* __restrict__ acc, const float* __restrict__ w,
                        const float* __restrict__ b, float* __restrict__ out) {
    int p = threadIdx.x;
    if (p < PP) {
        const float inv = 1.0f / ((float)NN * (float)NN * (float)SS);
        out[p] = acc[p] * inv * w[0] + b[0];
    }
}

extern "C" void kernel_launch(void* const* d_in, const int* in_sizes, int n_in,
                              void* d_out, int out_size, void* d_ws, size_t ws_size,
                              hipStream_t stream) {
    const float* hid    = (const float*)d_in[0];  // [768,64]
    const float* coords = (const float*)d_in[1];  // [4,768,3]
    const float* W_rbf  = (const float*)d_in[2];  // [64,64,32]
    const float* b_rbf  = (const float*)d_in[3];  // [32]
    const float* weight = (const float*)d_in[4];  // [1]
    const float* bias   = (const float*)d_in[5];  // [1]

    char* ws = (char*)d_ws;
    _Float16* hid_h = (_Float16*)ws;                              // 96 KB
    _Float16* T_h   = (_Float16*)(ws + 98304);                    // 3 MB
    float*    accb  = (float*)(ws + 98304 + 3145728);             // 16 B

    k_cvt_hid<<<(NN * DD + 255) / 256, 256, 0, stream>>>(hid, hid_h);
    k_calc_T<<<(NN * DD * SS) / 256, 256, 0, stream>>>(hid, W_rbf, T_h);
    k_zero<<<1, 32, 0, stream>>>(accb);
    k_fused<<<(NT * NT) / 8, 256, 0, stream>>>(T_h, hid_h, coords, b_rbf, accb);
    k_final<<<1, 32, 0, stream>>>(accb, weight, bias, (float*)d_out);
}
